// Hamiltonian_24326694764999
// MI455X (gfx1250) — matrix operand order repaired and checked by reading, not compiled, not run on hardware
//
#include <hip/hip_runtime.h>
#include <hip/hip_bf16.h>

typedef __attribute__((ext_vector_type(16))) _Float16 v16h;
typedef __attribute__((ext_vector_type(8)))  float    v8f;

#define S64 262144   // 64^3
#define S66 287496   // 66^3
#define S68 314432   // 68^3
#define S70 343000   // 70^3
#define CH  16

// ---------------------------------------------------------------------------
// Gradient bases: g(w) = 0.5*(s[w+1]-s[w-1]) on channel 1, 3-tap box over D/H.
// ---------------------------------------------------------------------------
__device__ __forceinline__ float gvalf(const float* s, int d, int h, int w) {
  if ((unsigned)d >= 64u || (unsigned)h >= 64u) return 0.0f;
  const int base = (d << 12) + (h << 6);
  float a = (w + 1 < 64) ? s[base + w + 1] : 0.0f;
  float b = (w >= 1)     ? s[base + w - 1] : 0.0f;
  return 0.5f * (a - b);
}

__global__ __launch_bounds__(256) void grad_kernel(const float* __restrict__ p,
                                                   const float* __restrict__ q,
                                                   float* __restrict__ gb) {
  int idx = blockIdx.x * 256 + threadIdx.x;
  if (idx >= 2 * S64) return;
  int b   = idx >> 18;
  int pos = idx & (S64 - 1);
  int d = pos >> 12, h = (pos >> 6) & 63, w = pos & 63;
  const float* sp = p + (size_t)b * CH * S64 + S64;   // channel 1
  const float* sq = q + (size_t)b * CH * S64 + S64;
  float gp = gvalf(sp, d, h, w);
  float gq = gvalf(sq, d, h, w);
  gb[0 * 2 * S64 + idx] = gvalf(sp, d - 1, h, w) + gp + gvalf(sp, d + 1, h, w); // dpx
  gb[1 * 2 * S64 + idx] = gvalf(sp, d, h - 1, w) + gp + gvalf(sp, d, h + 1, w); // dpy
  gb[2 * 2 * S64 + idx] = gvalf(sq, d - 1, h, w) + gq + gvalf(sq, d + 1, h, w); // dqx
  gb[3 * 2 * S64 + idx] = gvalf(sq, d, h - 1, w) + gq + gvalf(sq, d, h + 1, w); // dqy
}

// ---------------------------------------------------------------------------
// Stream the 6 big output blocks (p copy, dpx/dpy broadcast, q, dqx, dqy).
// float4 path: this is the dominant HBM traffic (~200 MB of stores).
// ---------------------------------------------------------------------------
__global__ __launch_bounds__(256) void assemble_kernel(const float* __restrict__ p,
                                                       const float* __restrict__ q,
                                                       const float* __restrict__ gb,
                                                       float* __restrict__ out) {
  const int E4 = 2 * CH * S64 / 4;          // 2097152 float4 per region (2^21)
  int gid = blockIdx.x * 256 + threadIdx.x;
  if (gid >= 6 * E4) return;
  int r = gid >> 21;                        // region 0..5
  int i = gid & (E4 - 1);
  float4 v;
  if (r == 0)      v = ((const float4*)p)[i];
  else if (r == 3) v = ((const float4*)q)[i];
  else {                                    // gradient broadcast across 16 chans
    int gi   = (r == 1) ? 0 : (r == 2) ? 1 : (r == 4) ? 2 : 3;
    int b    = i >> 20;                     // / (16*65536)
    int pos4 = i & 65535;
    v = ((const float4*)gb)[gi * 131072 + b * 65536 + pos4];
  }
  ((float4*)(out + 2 * S70))[(size_t)r * E4 + i] = v;
}

// ---------------------------------------------------------------------------
// conv1: out[o] = relu( Weff[o, 0..36] . feat[0..36] ), K padded to 64.
// feat = [ p(16) | q(16) | dpx dpy dqx dqy | 1 | 0.. ]
// ---------------------------------------------------------------------------
__device__ __forceinline__ float weff1(const float* W1, const float* b1, int m, int k) {
  if (k < 16) return W1[m * 96 + k];             // p channels
  if (k < 32) return W1[m * 96 + 48 + (k - 16)]; // q channels
  if (k < 36) {                                  // collapsed broadcast-grad cols
    int base = (k == 32) ? 16 : (k == 33) ? 32 : (k == 34) ? 64 : 80;
    float s = 0.f;
    for (int j = 0; j < 16; ++j) s += W1[m * 96 + base + j];
    return s;
  }
  if (k == 36) return b1[m];                     // bias via ones-channel
  return 0.f;
}

__global__ __launch_bounds__(256) void conv1_kernel(const float* __restrict__ p,
                                                    const float* __restrict__ q,
                                                    const float* __restrict__ W1,
                                                    const float* __restrict__ b1,
                                                    const float* __restrict__ gb,
                                                    float* __restrict__ h1) {
  const int lane = threadIdx.x & 31;
  const int n  = lane & 15;   // A row M == B column N for this lane
  const int hi = lane >> 4;   // lane half selects K sub-range
  const int waveId = blockIdx.x * 8 + (threadIdx.x >> 5);
  const int nWaves = gridDim.x * 8;

  // A matrix (weights, constant per kernel): contiguous K order, k = 16*hi + e (same as B)
  v16h a0, a1;
  for (int e = 0; e < 16; ++e) {
    int kc = 16 * hi + e;
    a0[e] = (_Float16)weff1(W1, b1, n, kc);
    a1[e] = (_Float16)weff1(W1, b1, n, kc + 32);
  }

  const int TPB = (S66 + 15) / 16;  // 17969 tiles per batch
  for (int t = waveId; t < 2 * TPB; t += nWaves) {
    int b  = (t >= TPB);
    int ti = t - b * TPB;
    int fp = ti * 16 + n;                   // flat pos in 66^3 grid
    bool valid = fp < S66;
    int cfp = valid ? fp : (S66 - 1);
    int d = cfp / 4356;
    int rem = cfp - d * 4356;
    int h = rem / 66;
    int w = rem - h * 66;
    int sd = d - 1, sh = h - 1, sw = w - 1; // source coords in 64^3
    bool inr = valid && ((unsigned)sd < 64u) && ((unsigned)sh < 64u) && ((unsigned)sw < 64u);
    int csd = sd < 0 ? 0 : (sd > 63 ? 63 : sd);
    int csh = sh < 0 ? 0 : (sh > 63 ? 63 : sh);
    int csw = sw < 0 ? 0 : (sw > 63 ? 63 : sw);
    int spos = (csd << 12) + (csh << 6) + csw;
    float msk = inr ? 1.0f : 0.0f;

    // B chunk 0: K 0..15 = p channels (lanes 0-15), K 16..31 = q (lanes 16-31)
    const float* base = (hi == 0 ? p : q) + (size_t)b * CH * S64 + spos;
    v16h bb0, bb1;
    for (int e = 0; e < 16; ++e)
      bb0[e] = (_Float16)(base[(size_t)e * S64] * msk);

    // B chunk 1: K 32..35 = gradients, K 36 = ones; lanes 16-31 (K 48..63) = 0
    float hm = (hi == 0) ? 1.0f : 0.0f;
    int gp = b * S64 + spos;
    bb1[0] = (_Float16)(gb[0 * 2 * S64 + gp] * msk * hm);
    bb1[1] = (_Float16)(gb[1 * 2 * S64 + gp] * msk * hm);
    bb1[2] = (_Float16)(gb[2 * 2 * S64 + gp] * msk * hm);
    bb1[3] = (_Float16)(gb[3 * 2 * S64 + gp] * msk * hm);
    bb1[4] = (_Float16)((valid ? 1.0f : 0.0f) * hm);
    for (int e = 5; e < 16; ++e) bb1[e] = (_Float16)0.0f;

    v8f acc = {};
    acc = __builtin_amdgcn_wmma_f32_16x16x32_f16(false, a0, false, bb0, (short)0, acc, false, false);
    acc = __builtin_amdgcn_wmma_f32_16x16x32_f16(false, a1, false, bb1, (short)0, acc, false, false);

    if (valid) {  // C layout: VGPR i -> row i + 8*hi, col n
      size_t ob = ((size_t)b * CH + 8 * hi) * S66 + fp;
      for (int i = 0; i < 8; ++i) {
        float v = acc[i];
        h1[ob + (size_t)i * S66] = v > 0.f ? v : 0.f;
      }
    }
  }
}

// ---------------------------------------------------------------------------
// conv2: K = 16 channels + ones(b2), padded to 32 -> single WMMA per tile.
// ---------------------------------------------------------------------------
__global__ __launch_bounds__(256) void conv2_kernel(const float* __restrict__ h1,
                                                    const float* __restrict__ W2,
                                                    const float* __restrict__ b2,
                                                    float* __restrict__ h2) {
  const int lane = threadIdx.x & 31;
  const int n  = lane & 15;
  const int hi = lane >> 4;
  const int waveId = blockIdx.x * 8 + (threadIdx.x >> 5);
  const int nWaves = gridDim.x * 8;

  v16h a0;
  for (int e = 0; e < 16; ++e) {
    int kc = 16 * hi + e;
    float wv = (kc < 16) ? W2[n * 16 + kc] : (kc == 16 ? b2[n] : 0.0f);
    a0[e] = (_Float16)wv;
  }

  const int TPB = S68 / 16;  // 19652 (exact)
  for (int t = waveId; t < 2 * TPB; t += nWaves) {
    int b  = (t >= TPB);
    int ti = t - b * TPB;
    int fp = ti * 16 + n;
    int d = fp / 4624;
    int rem = fp - d * 4624;
    int h = rem / 68;
    int w = rem - h * 68;
    int sd = d - 1, sh = h - 1, sw = w - 1;   // source in 66^3
    bool inr = ((unsigned)sd < 66u) && ((unsigned)sh < 66u) && ((unsigned)sw < 66u);
    int csd = sd < 0 ? 0 : (sd > 65 ? 65 : sd);
    int csh = sh < 0 ? 0 : (sh > 65 ? 65 : sh);
    int csw = sw < 0 ? 0 : (sw > 65 ? 65 : sw);
    int spos = csd * 4356 + csh * 66 + csw;
    float msk = inr ? 1.0f : 0.0f;

    const float* hb = h1 + (size_t)b * CH * S66 + spos;
    v16h bb;
    for (int e = 0; e < 16; ++e) {
      float v = (hi == 0) ? (hb[(size_t)e * S66] * msk)   // K = e : channels
                          : ((e == 0) ? 1.0f : 0.0f);     // K = 16 : ones
      bb[e] = (_Float16)v;
    }

    v8f acc = {};
    acc = __builtin_amdgcn_wmma_f32_16x16x32_f16(false, a0, false, bb, (short)0, acc, false, false);

    size_t ob = ((size_t)b * CH + 8 * hi) * S68 + fp;
    for (int i = 0; i < 8; ++i) {
      float v = acc[i];
      h2[ob + (size_t)i * S68] = v > 0.f ? v : 0.f;
    }
  }
}

// ---------------------------------------------------------------------------
// conv3: single output channel -> plain VALU dot over 16 chans (L2-resident).
// ---------------------------------------------------------------------------
__global__ __launch_bounds__(256) void conv3_kernel(const float* __restrict__ h2,
                                                    const float* __restrict__ W3,
                                                    const float* __restrict__ b3,
                                                    float* __restrict__ out) {
  int idx = blockIdx.x * 256 + threadIdx.x;
  if (idx >= 2 * S70) return;
  int b = idx / S70;
  int pos = idx - b * S70;
  int d = pos / 4900;
  int rem = pos - d * 4900;
  int h = rem / 70;
  int w = rem - h * 70;
  int sd = d - 1, sh = h - 1, sw = w - 1;   // source in 68^3
  float acc = b3[0];
  if ((unsigned)sd < 68u && (unsigned)sh < 68u && (unsigned)sw < 68u) {
    int spos = sd * 4624 + sh * 68 + sw;
    const float* hb = h2 + (size_t)b * CH * S68 + spos;
    #pragma unroll
    for (int c = 0; c < 16; ++c) acc += W3[c] * hb[(size_t)c * S68];
  }
  out[idx] = acc > 0.f ? acc : 0.f;
}

// ---------------------------------------------------------------------------
extern "C" void kernel_launch(void* const* d_in, const int* in_sizes, int n_in,
                              void* d_out, int out_size, void* d_ws, size_t ws_size,
                              hipStream_t stream) {
  const float* p  = (const float*)d_in[0];
  const float* q  = (const float*)d_in[1];
  const float* W1 = (const float*)d_in[2];
  const float* b1 = (const float*)d_in[3];
  const float* W2 = (const float*)d_in[4];
  const float* b2 = (const float*)d_in[5];
  const float* W3 = (const float*)d_in[6];
  const float* b3 = (const float*)d_in[7];
  float* out = (float*)d_out;
  float* ws  = (float*)d_ws;

  // Workspace layout (floats): 4 grad bases | h1 [2,16,66^3] | h2 [2,16,68^3]
  float* gb = ws;                          // 4 * 2*S64      = 2,097,152
  float* h1 = gb + 4 * 2 * S64;            // 2*16*S66       = 9,199,872
  float* h2 = h1 + (size_t)2 * CH * S66;   // 2*16*S68       = 10,061,824  (~85 MB total)

  grad_kernel    <<<(2 * S64 + 255) / 256, 256, 0, stream>>>(p, q, gb);
  assemble_kernel<<<(6 * (2 * CH * S64 / 4) + 255) / 256, 256, 0, stream>>>(p, q, gb, out);
  conv1_kernel   <<<(2 * ((S66 + 15) / 16) + 7) / 8, 256, 0, stream>>>(p, q, W1, b1, gb, h1);
  conv2_kernel   <<<(2 * (S68 / 16) + 7) / 8, 256, 0, stream>>>(h1, W2, b2, h2);
  conv3_kernel   <<<(2 * S70 + 255) / 256, 256, 0, stream>>>(h2, W3, b3, out);
}
